// CrossAttentionFusionHead_63986422776408
// MI455X (gfx1250) — matrix operand order repaired and checked by reading, not compiled, not run on hardware
//
#include <hip/hip_runtime.h>

#define E 512
#define H 256
#define NC 5
#define ROWS 64            // rows per block: 4 M-tiles -> 4x B-fragment reuse
#define NTHREADS 256       // 8 waves

typedef __bf16 bf16_t;
typedef __attribute__((ext_vector_type(16))) __bf16 v16bf;
typedef __attribute__((ext_vector_type(8)))  float  v8f;

// ---------------- workspace layout ----------------
// bf16 region (element offsets):
#define WS_WIMG   0         // [256][512]
#define WS_WTXT   131072    // [256][512]
#define WS_WCOMB1 262144    // [256][256]  Wo_t2i @ Wv_t2i   (attention is linear: softmax(1)=1)
#define WS_WCOMB2 327680    // [256][256]  Wo_i2t @ Wv_i2t
#define WS_WC1    393216    // [256][512]
#define WS_WC2    524288    // [16][256] (rows 5..15 zero-padded)
#define WS_BF16_END 528384  // bf16 elems -> 1056768 bytes
// f32 region (byte offset / float index):
#define WS_BCOMB1_BYTES 1056768   // 256 f32 : Wo_t2i@bv + bo
#define WS_BCOMB2_BYTES 1057792   // 256 f32

// ---------------- LDS layout (byte offsets), total 198656 B --------------
// sXimg  [64][512] bf16 @ 0       (64K) -+ overlaid by sFusedF [64][512] f32 (128K)
// sXtxt  [64][512] bf16 @ 65536   (64K) -+ then by sH [64][256] bf16 (32K)
// sImgBf [64][256] bf16 @ 131072  (32K) -+ overlaid by sFusedBf [64][512] bf16
// sTxtBf [64][256] bf16 @ 163840  (32K) -+ (dead after GEMM4)
// red    512 f32        @ 196608  ( 2K)
#define SMEM_BYTES 198656

union FragU { v16bf v; uint4 u[2]; };

// A fragment: 16x32 bf16 from row-major LDS tile [16][ldk].
// Lane m (m<16): row m, K = k0..k0+7 then k0+16..k0+23 ; lane m+16: K +8 / +24.
__device__ __forceinline__ v16bf load_frag_a(const bf16_t* A, int ldk, int k0, int lane) {
  const int row = lane & 15, hi = lane >> 4;
  const bf16_t* p = A + row * ldk + k0 + hi * 8;
  FragU f;
  f.u[0] = *(const uint4*)(p);
  f.u[1] = *(const uint4*)(p + 16);
  return f.v;
}

// B fragment: 32x16 (B[k][n] = W[n][k], W row-major [N][ldk]).
// Lane n (n<16): column n, K = k0..k0+7 then k0+16..k0+23 ; lane n+16: K +8 / +24.
__device__ __forceinline__ v16bf load_frag_b(const bf16_t* W, int ldk, int n0, int k0, int lane) {
  const int col = lane & 15, hi = lane >> 4;
  const bf16_t* p = W + (size_t)(n0 + col) * ldk + k0 + hi * 8;
  FragU f;
  f.u[0] = *(const uint4*)(p);
  f.u[1] = *(const uint4*)(p + 16);
  return f.v;
}

__device__ __forceinline__ float gelu_exact(float x) {
  return 0.5f * x * (1.0f + erff(x * 0.70710678118654752f));
}

// 64x256 GEMM: A[64][KTOT] (LDS bf16) x W[256][KTOT]^T (global bf16).
// 8 waves; wave w owns N-tiles {2w, 2w+1} x M-tiles {0..3} (8 accumulators).
// Both A (LDS) and B (global) fragments are double-buffered: k+1's 12 loads are
// in flight while k's 8 WMMAs issue, so no ds/load waits sit between WMMAs.
// EP: 1 = bias -> bf16 ; 2 = bias + bf16 residual -> f32 (ld 512) ; 3 = bias + gelu -> bf16
template<int KTOT, int EP>
__device__ __forceinline__ void gemm_block(const bf16_t* A, const bf16_t* W, const float* bias,
                                           float* outF, const bf16_t* residB, bf16_t* outB,
                                           int tid)
{
  const int lane = tid & 31;
  const int w    = tid >> 5;              // 0..7
  const int col  = lane & 15, hi = lane >> 4;
  const int nb   = w * 32;                // 8 waves x 2 N-tiles x 16 = 256
  v8f acc[2][4] = {};
  v16bf a[2][4], bfr[2][2];
  bfr[0][0] = load_frag_b(W, KTOT, nb,      0, lane);
  bfr[0][1] = load_frag_b(W, KTOT, nb + 16, 0, lane);
#pragma unroll
  for (int m = 0; m < 4; ++m)
    a[0][m] = load_frag_a(A + m * 16 * KTOT, KTOT, 0, lane);
#pragma unroll
  for (int ks = 0; ks < KTOT / 32; ++ks) {
    const int k0  = ks * 32;
    const int cur = ks & 1, nxt = cur ^ 1;
    if (k0 + 32 < KTOT) {                 // prefetch next k-step's A+B frags
      bfr[nxt][0] = load_frag_b(W, KTOT, nb,      k0 + 32, lane);
      bfr[nxt][1] = load_frag_b(W, KTOT, nb + 16, k0 + 32, lane);
#pragma unroll
      for (int m = 0; m < 4; ++m)
        a[nxt][m] = load_frag_a(A + m * 16 * KTOT, KTOT, k0 + 32, lane);
    }
#pragma unroll
    for (int n = 0; n < 2; ++n)
#pragma unroll
      for (int m = 0; m < 4; ++m)
        acc[n][m] = __builtin_amdgcn_wmma_f32_16x16x32_bf16(false, a[cur][m], false, bfr[cur][n],
                                                            (short)0, acc[n][m], false, false);
  }
#pragma unroll
  for (int n = 0; n < 2; ++n) {
    const int c  = nb + n * 16 + col;
    const float bv = bias[c];
#pragma unroll
    for (int m = 0; m < 4; ++m) {
#pragma unroll
      for (int r = 0; r < 8; ++r) {
        const int row = m * 16 + r + 8 * hi;   // C/D: lanes 0-15 -> M=r, 16-31 -> M=r+8
        float v = acc[n][m][r] + bv;
        if (EP == 2) v += (float)residB[row * H + c];
        if (EP == 3) v = gelu_exact(v);
        if (EP == 2) outF[row * (2 * H) + c] = v;
        else         outB[row * H + c] = (bf16_t)v;
      }
    }
  }
}

// Cooperative LayerNorm over 64 rows x Wd cols (4 threads per row, 256 threads).
// outB==nullptr -> in-place f32; else write bf16 [64][512] at column offset c0.
__device__ __forceinline__ void block_layernorm(float* buf, int c0, int Wd,
                                                const float* g, const float* b,
                                                float* red, bf16_t* outB, int tid)
{
  const int row = tid >> 2, seg = tid & 3;
  const int cpt = Wd >> 2;
  float* x = buf + row * (2 * H) + c0;
  float s = 0.f, s2 = 0.f;
  for (int j = 0; j < cpt; ++j) { float v = x[seg * cpt + j]; s += v; s2 += v * v; }
  red[row * 4 + seg] = s;
  red[256 + row * 4 + seg] = s2;
  __syncthreads();
  float ts = 0.f, ts2 = 0.f;
  for (int j = 0; j < 4; ++j) { ts += red[row * 4 + j]; ts2 += red[256 + row * 4 + j]; }
  const float mu   = ts / (float)Wd;
  const float rstd = rsqrtf(fmaxf(ts2 / (float)Wd - mu * mu, 0.f) + 1e-5f);
  for (int j = 0; j < cpt; ++j) {
    const int c = seg * cpt + j;
    const float y = (x[c] - mu) * rstd * g[c] + b[c];
    if (outB) outB[row * (2 * H) + c0 + c] = (bf16_t)y;
    else      x[c] = y;
  }
}

// Stage 64x512 fp32 tile from global -> bf16 LDS (contiguous, float4 loads).
__device__ __forceinline__ void stage_x(const float* X, bf16_t* dst, int tid) {
  const float4* src = (const float4*)X;
  for (int i = tid; i < (ROWS * E) / 4; i += NTHREADS) {
    float4 v = src[i];
    union { bf16_t h[4]; uint2 u; } o;
    o.h[0] = (bf16_t)v.x; o.h[1] = (bf16_t)v.y;
    o.h[2] = (bf16_t)v.z; o.h[3] = (bf16_t)v.w;
    *(uint2*)(dst + i * 4) = o.u;
  }
}

struct Params {
  const float* ximg; const float* xtxt;
  const bf16_t* wimg; const bf16_t* wtxt;
  const bf16_t* wcomb1; const bf16_t* wcomb2;
  const bf16_t* wc1; const bf16_t* wc2;
  const float* b_img; const float* b_txt;
  const float* bcomb1; const float* bcomb2;
  const float* n1g; const float* n1b;
  const float* n2g; const float* n2b;
  const float* n3g; const float* n3b;
  const float* b_c1; const float* b_c2;
  float* out; int B;
};

__global__ __launch_bounds__(NTHREADS)
void fused_cross_attn_head(Params p)
{
  extern __shared__ char smem[];
  bf16_t* sXimg    = (bf16_t*)(smem + 0);
  bf16_t* sXtxt    = (bf16_t*)(smem + 65536);
  float*  sFusedF  = (float*)(smem + 0);        // overlays sXimg/sXtxt (dead after GEMM1/2)
  bf16_t* sH       = (bf16_t*)(smem + 0);       // overlays sFusedF (dead after LN3)
  bf16_t* sImgBf   = (bf16_t*)(smem + 131072);
  bf16_t* sTxtBf   = (bf16_t*)(smem + 163840);
  bf16_t* sFusedBf = (bf16_t*)(smem + 131072);  // overlays sImgBf/sTxtBf (dead after GEMM4)
  float*  red      = (float*)(smem + 196608);

  const int tid = threadIdx.x;
  const size_t m0 = (size_t)blockIdx.x * ROWS;

  // stage features (fp32 -> bf16)
  stage_x(p.ximg + m0 * E, sXimg, tid);
  stage_x(p.xtxt + m0 * E, sXtxt, tid);
  __syncthreads();

  // img = Ximg @ Wimg.T + b ; txt = Xtxt @ Wtxt.T + b  (bf16; also the residuals)
  gemm_block<E, 1>(sXimg, p.wimg, p.b_img, nullptr, nullptr, sImgBf, tid);
  __syncthreads();
  gemm_block<E, 1>(sXtxt, p.wtxt, p.b_txt, nullptr, nullptr, sTxtBf, tid);
  __syncthreads();
  // attention with 1 key is linear: ta_pre = txt + img @ (Wo1·Wv1).T + (Wo1·bv1 + bo1)
  gemm_block<H, 2>(sImgBf, p.wcomb1, p.bcomb1, sFusedF, sTxtBf, nullptr, tid);
  __syncthreads();
  block_layernorm(sFusedF, 0, H, p.n1g, p.n1b, red, nullptr, tid);
  __syncthreads();
  // ia_pre = img + txt @ (Wo2·Wv2).T + (Wo2·bv2 + bo2)
  gemm_block<H, 2>(sTxtBf, p.wcomb2, p.bcomb2, sFusedF + H, sImgBf, nullptr, tid);
  __syncthreads();
  block_layernorm(sFusedF, H, H, p.n2g, p.n2b, red, nullptr, tid);
  __syncthreads();
  // cls LN over full 512 -> bf16 for classifier GEMM
  block_layernorm(sFusedF, 0, 2 * H, p.n3g, p.n3b, red, sFusedBf, tid);
  __syncthreads();
  // h = gelu(fused_ln @ Wc1.T + b)
  gemm_block<E, 3>(sFusedBf, p.wc1, p.b_c1, nullptr, nullptr, sH, tid);
  __syncthreads();

  // logits = h @ Wc2p.T + b : 4 M-tiles x 1 N-tile; waves 0..3, one M-tile each
  {
    const int lane = tid & 31;
    const int w    = tid >> 5;
    if (w < 4) {
      const int col = lane & 15, hi = lane >> 4;
      v8f acc = {};
      const bf16_t* Ah = sH + w * 16 * H;
#pragma unroll
      for (int ks = 0; ks < H / 32; ++ks) {
        v16bf av = load_frag_a(Ah, H, ks * 32, lane);
        v16bf bv = load_frag_b(p.wc2, H, 0, ks * 32, lane);
        acc = __builtin_amdgcn_wmma_f32_16x16x32_bf16(false, av, false, bv,
                                                      (short)0, acc, false, false);
      }
      if (col < NC) {
        const float bb = p.b_c2[col];
#pragma unroll
        for (int r = 0; r < 8; ++r) {
          const size_t row = m0 + (size_t)(w * 16 + r + 8 * hi);
          p.out[row * NC + col] = acc[r] + bb;
        }
      }
    }
    // softmax over a single key => attention weights are exactly 1.0
    if (tid < ROWS) {
      const size_t b5 = (size_t)p.B * NC;
      p.out[b5 + m0 + tid]               = 1.0f;  // txt_weights
      p.out[b5 + (size_t)p.B + m0 + tid] = 1.0f;  // img_weights
    }
  }
}

// ------------------------- weight prep kernels ---------------------------
__global__ void cvt_bf16_kernel(const float* __restrict__ s, bf16_t* __restrict__ d, int n) {
  int i = blockIdx.x * blockDim.x + threadIdx.x;
  if (i < n) d[i] = (bf16_t)s[i];
}

__global__ void cvt_pad_w2_kernel(const float* __restrict__ s, bf16_t* __restrict__ d) {
  int i = blockIdx.x * blockDim.x + threadIdx.x;
  if (i < 16 * H) {
    int r = i / H, c = i % H;
    d[i] = (r < NC) ? (bf16_t)s[r * H + c] : (bf16_t)0.0f;
  }
}

// Wcomb[n][k] = sum_j Wo[n][j] * Wv[j][k], fp32 accumulate, bf16 store.
// One block (256 threads) per output row n -> Wo[n][j] is a scalar broadcast,
// Wv[j*256+k] loads are coalesced across lanes.
__global__ void comb_w_kernel(const float* __restrict__ Wo, const float* __restrict__ Wv,
                              bf16_t* __restrict__ dst) {
  const int n = blockIdx.x, k = threadIdx.x;
  float acc = 0.f;
  for (int j = 0; j < H; ++j) acc += Wo[n * H + j] * Wv[j * H + k];
  dst[n * H + k] = (bf16_t)acc;
}

// bcomb[n] = sum_j Wo[n][j] * bv[j] + bo[n]
__global__ void comb_b_kernel(const float* __restrict__ Wo, const float* __restrict__ bv,
                              const float* __restrict__ bo, float* __restrict__ dst) {
  const int n = blockIdx.x * blockDim.x + threadIdx.x;
  if (n < H) {
    float acc = bo[n];
    for (int j = 0; j < H; ++j) acc += Wo[n * H + j] * bv[j];
    dst[n] = acc;
  }
}

extern "C" void kernel_launch(void* const* d_in, const int* in_sizes, int n_in,
                              void* d_out, int out_size, void* d_ws, size_t ws_size,
                              hipStream_t stream) {
  (void)n_in; (void)out_size; (void)ws_size;
  bf16_t* ws = (bf16_t*)d_ws;
  float*  bcomb1 = (float*)((char*)d_ws + WS_BCOMB1_BYTES);
  float*  bcomb2 = (float*)((char*)d_ws + WS_BCOMB2_BYTES);
  const int B = in_sizes[0] / E;

  auto cvt = [&](const void* src, size_t off, int n) {
    cvt_bf16_kernel<<<(n + 255) / 256, 256, 0, stream>>>((const float*)src, ws + off, n);
  };
  cvt(d_in[2],  WS_WIMG, H * E);       // img_proj_w
  cvt(d_in[4],  WS_WTXT, H * E);       // txt_proj_w
  cvt(d_in[20], WS_WC1,  H * 2 * H);   // cls_w1
  cvt_pad_w2_kernel<<<(16 * H + 255) / 256, 256, 0, stream>>>((const float*)d_in[22], ws + WS_WC2);
  // fold the (linear) 1-key attention: Wcomb = Wo @ Wv ; bcomb = Wo @ bv + bo
  comb_w_kernel<<<H, H, 0, stream>>>((const float*)d_in[8],
                                     (const float*)d_in[6] + 2 * H * H, ws + WS_WCOMB1);
  comb_b_kernel<<<1, H, 0, stream>>>((const float*)d_in[8],
                                     (const float*)d_in[7] + 2 * H,
                                     (const float*)d_in[9], bcomb1);
  comb_w_kernel<<<H, H, 0, stream>>>((const float*)d_in[12],
                                     (const float*)d_in[10] + 2 * H * H, ws + WS_WCOMB2);
  comb_b_kernel<<<1, H, 0, stream>>>((const float*)d_in[12],
                                     (const float*)d_in[11] + 2 * H,
                                     (const float*)d_in[13], bcomb2);

  Params p;
  p.ximg = (const float*)d_in[0];
  p.xtxt = (const float*)d_in[1];
  p.wimg   = ws + WS_WIMG;   p.wtxt   = ws + WS_WTXT;
  p.wcomb1 = ws + WS_WCOMB1; p.wcomb2 = ws + WS_WCOMB2;
  p.wc1    = ws + WS_WC1;    p.wc2    = ws + WS_WC2;
  p.b_img = (const float*)d_in[3];
  p.b_txt = (const float*)d_in[5];
  p.bcomb1 = bcomb1;
  p.bcomb2 = bcomb2;
  p.n1g = (const float*)d_in[14]; p.n1b = (const float*)d_in[15];
  p.n2g = (const float*)d_in[16]; p.n2b = (const float*)d_in[17];
  p.n3g = (const float*)d_in[18]; p.n3b = (const float*)d_in[19];
  p.b_c1 = (const float*)d_in[21];
  p.b_c2 = (const float*)d_in[23];
  p.out = (float*)d_out;
  p.B = B;

  // 194 KB dynamic LDS per block (one block per WGP; WGP has 320 KB LDS)
  hipFuncSetAttribute((const void*)fused_cross_attn_head,
                      hipFuncAttributeMaxDynamicSharedMemorySize, SMEM_BYTES);
  fused_cross_attn_head<<<B / ROWS, NTHREADS, SMEM_BYTES, stream>>>(p);
}
